// GMA_37546604102396
// MI455X (gfx1250) — matrix operand order repaired and checked by reading, not compiled, not run on hardware
//
#include <hip/hip_runtime.h>
#include <hip/hip_bf16.h>

typedef __attribute__((ext_vector_type(16))) _Float16 v16h;
typedef __attribute__((ext_vector_type(8)))  float    v8f;
typedef __attribute__((ext_vector_type(8)))  unsigned v8u;

union FragCast { v8u u; v16h h; };

__device__ inline unsigned pack2h(float x, float y) {
    union { unsigned u; _Float16 h[2]; } c;
    c.h[0] = (_Float16)x; c.h[1] = (_Float16)y;
    return c.u;
}

// ---------------- small utility kernels ----------------

__global__ __launch_bounds__(256) void zero_f32(float* __restrict__ p, long n) {
    long i = (long)blockIdx.x * blockDim.x + threadIdx.x;
    if (i < n) p[i] = 0.f;
}

__global__ __launch_bounds__(256) void init_deg(float* __restrict__ deg, int n) {
    int i = blockIdx.x * blockDim.x + threadIdx.x;
    if (i < n) deg[i] = 1.f;   // self loop
}

__global__ __launch_bounds__(256) void count_deg(const int* __restrict__ dst,
                                                 float* __restrict__ deg, int E) {
    int i = blockIdx.x * blockDim.x + threadIdx.x;
    if (i < E) atomicAdd(&deg[dst[i]], 1.f);
}

__global__ __launch_bounds__(256) void finalize_deg(const float* __restrict__ deg,
                                                    float* __restrict__ dinv,
                                                    float* __restrict__ invd, int n) {
    int i = blockIdx.x * blockDim.x + threadIdx.x;
    if (i < n) {
        float d = deg[i];
        dinv[i] = rsqrtf(d);
        invd[i] = 1.f / d;
    }
}

// ---------------- GCN message passing ----------------

// one wave (32 lanes) per edge; each lane handles 4 channels
__global__ __launch_bounds__(256) void gcn_scatter(const float* __restrict__ hlin,
                                                   const int* __restrict__ src,
                                                   const int* __restrict__ dst,
                                                   const float* __restrict__ dinv,
                                                   float* __restrict__ agg, int E) {
    long tid = (long)blockIdx.x * blockDim.x + threadIdx.x;
    int e = (int)(tid >> 5);
    if (e >= E) return;
    int s = src[e], d = dst[e];
    int c = ((int)tid & 31) * 4;
    const float* hp = hlin + (long)s * 128 + c;
    __builtin_prefetch(hp, 0, 1);            // gfx1250 global_prefetch path
    float w = dinv[s] * dinv[d];
    const float4 hv = *reinterpret_cast<const float4*>(hp);
    float* out = agg + (long)d * 128 + c;
    atomicAdd(out + 0, hv.x * w);
    atomicAdd(out + 1, hv.y * w);
    atomicAdd(out + 2, hv.z * w);
    atomicAdd(out + 3, hv.w * w);
}

// agg[n][c] = (relu?)(agg + hlin*invdeg[n] + bias[c])
__global__ __launch_bounds__(256) void gcn_finalize(float* __restrict__ agg,
                                                    const float* __restrict__ hlin,
                                                    const float* __restrict__ invdeg,
                                                    const float* __restrict__ bias,
                                                    int relu, long total) {
    long i = (long)blockIdx.x * blockDim.x + threadIdx.x;
    if (i >= total) return;
    int n = (int)(i >> 7), c = (int)(i & 127);
    float r = agg[i] + hlin[i] * invdeg[n] + bias[c];
    if (relu) r = fmaxf(r, 0.f);
    agg[i] = r;
}

// ---------------- WMMA fragment loaders (contiguous k = 16*half + s order) ----------------
// LDS tiles are stored as packed f16 pairs in unsigned arrays (row pitch 16 uints = 32 f16)

// A 16x32 f16: lanes0-15 = rows, pair p at kb = half*16 + 2p
__device__ inline v16h load_a_lds(const unsigned* sA, int rowBase, int lane) {
    int m = rowBase + (lane & 15);
    int half = lane >> 4;
    const unsigned* p = sA + m * 16;
    FragCast f;
#pragma unroll
    for (int q = 0; q < 8; ++q) {
        int idx = half * 8 + q;
        f.u[q] = p[idx];
    }
    return f.h;
}

// B 32x16 f16 stored transposed as [n][k-pairs]: pair p at kb = half*16 + 2p
__device__ inline v16h load_b_lds(const unsigned* sB, int nBase, int lane) {
    int n = nBase + (lane & 15);
    int half = lane >> 4;
    const unsigned* p = sB + n * 16 + half * 8;
    FragCast f;
#pragma unroll
    for (int q = 0; q < 8; ++q) f.u[q] = p[q];
    return f.h;
}

// ---------------- generic GEMM: C[M,128] = epi(A[M,K] @ B[K,128]) ----------------
// mode 0: acc (+bias if bias!=null); mode 1: relu(acc+bias); mode 2: skip + relu(acc+bias)
__global__ __launch_bounds__(256) void gemm128(const float* __restrict__ A,
                                               const float* __restrict__ B,
                                               const float* __restrict__ bias,
                                               const float* __restrict__ skip,
                                               float* __restrict__ C,
                                               int M, int K, int mode) {
    __shared__ unsigned sAu[64 * 16];    // 64 rows x 32 f16 (packed pairs)
    __shared__ unsigned sBu[128 * 16];   // transposed [n][32 f16]
    int tid = threadIdx.x;
    int wave = tid >> 5, lane = tid & 31;
    int m0 = blockIdx.x * 64;
    int rt = wave >> 1;      // row tile 0..3
    int cg = wave & 1;       // col half

    v8f acc[4];
#pragma unroll
    for (int i = 0; i < 4; ++i)
#pragma unroll
        for (int v = 0; v < 8; ++v) acc[i][v] = 0.f;

    for (int kk = 0; kk < K; kk += 32) {
        // stage A: 64x32 floats = 512 float4; clamp row (tail rows are dont-care)
#pragma unroll
        for (int i = 0; i < 2; ++i) {
            int idx4 = i * 256 + tid;          // 0..511
            int r = idx4 >> 3, c4 = (idx4 & 7) * 4;
            int gm = m0 + r;
            gm = (gm < M) ? gm : (M - 1);
            const float4 v = *(const float4*)(A + (long)gm * K + kk + c4);
            uint2 pk;
            pk.x = pack2h(v.x, v.y);
            pk.y = pack2h(v.z, v.w);
            *(uint2*)(sAu + r * 16 + (c4 >> 1)) = pk;
        }
        // stage B^T: 2048 k-pairs; pack {B[k][n], B[k+1][n]}
#pragma unroll
        for (int i = 0; i < 8; ++i) {
            int idx = i * 256 + tid;           // 0..2047
            int n = idx & 127, kp = idx >> 7;  // kp 0..15
            const float* bp = B + (long)(kk + 2 * kp) * 128 + n;
            sBu[n * 16 + kp] = pack2h(bp[0], bp[128]);
        }
        __syncthreads();

        v16h a = load_a_lds(sAu, rt * 16, lane);
#pragma unroll
        for (int i = 0; i < 4; ++i) {
            v16h b = load_b_lds(sBu, (cg * 4 + i) * 16, lane);
            acc[i] = __builtin_amdgcn_wmma_f32_16x16x32_f16(
                false, a, false, b, (short)0, acc[i], false, false);
        }
        __syncthreads();
    }

    int n_l = lane & 15, half = lane >> 4;
#pragma unroll
    for (int i = 0; i < 4; ++i) {
        int n = (cg * 4 + i) * 16 + n_l;
        float bv = bias ? bias[n] : 0.f;
#pragma unroll
        for (int v = 0; v < 8; ++v) {
            int m = m0 + rt * 16 + half * 8 + v;
            if (m < M) {
                float r = acc[i][v] + bv;
                if (mode >= 1) r = fmaxf(r, 0.f);
                if (mode == 2) r += skip[(long)m * 128 + n];
                C[(long)m * 128 + n] = r;
            }
        }
    }
}

// ---------------- attention: scores = (Qh . Kh^T)/sqrt(128), one WMMA per tile ----------------
// grid: (q_tiles*k_tiles, 4 heads, 128 graphs), 32 threads
__global__ __launch_bounds__(32) void attn_scores(const float* __restrict__ Q,
                                                  const float* __restrict__ K,
                                                  _Float16* __restrict__ S,
                                                  int nq_pad, int L_pad,
                                                  int q_rpb, int k_rpb, int k_tiles) {
    int b = blockIdx.z, h = blockIdx.y;
    int qt = blockIdx.x / k_tiles, kt = blockIdx.x % k_tiles;
    int lane = threadIdx.x;
    int half = lane >> 4, l15 = lane & 15;

    long qrow = (long)(q_rpb ? b * q_rpb : 0) + qt * 16 + l15;
    const float* qp = Q + qrow * 128 + h * 32;
    FragCast fa;
#pragma unroll
    for (int p = 0; p < 8; ++p) {
        int kb = half * 16 + 2 * p;
        float2 f = *(const float2*)(qp + kb);
        fa.u[p] = pack2h(f.x, f.y);
    }
    long krow = (long)b * k_rpb + kt * 16 + l15;
    const float* kp = K + krow * 128 + h * 32;
    FragCast fb;
#pragma unroll
    for (int p = 0; p < 8; ++p) {
        int kb = half * 16 + 2 * p;
        float2 f = *(const float2*)(kp + kb);
        fb.u[p] = pack2h(f.x, f.y);
    }
    v8f acc;
#pragma unroll
    for (int v = 0; v < 8; ++v) acc[v] = 0.f;
    acc = __builtin_amdgcn_wmma_f32_16x16x32_f16(false, fa.h, false, fb.h,
                                                 (short)0, acc, false, false);
    const float scale = 0.08838834764831845f;   // 1/sqrt(128)
    long sbase = ((long)(b * 4 + h)) * nq_pad * L_pad;
    int n = kt * 16 + l15;
#pragma unroll
    for (int v = 0; v < 8; ++v) {
        int m = qt * 16 + half * 8 + v;
        S[sbase + (long)m * L_pad + n] = (_Float16)(acc[v] * scale);
    }
}

// one wave per (graph, head, query) row; zero the padded tail [L, L_pad)
__global__ __launch_bounds__(32) void attn_softmax(_Float16* __restrict__ S,
                                                   int nq_pad, int L, int L_pad) {
    int b = blockIdx.z, h = blockIdx.y, q = blockIdx.x;
    int lane = threadIdx.x;
    _Float16* row = S + (((long)(b * 4 + h)) * nq_pad + q) * L_pad;
    float m = -1e30f;
    for (int k = lane; k < L; k += 32) m = fmaxf(m, (float)row[k]);
#pragma unroll
    for (int o = 16; o; o >>= 1) m = fmaxf(m, __shfl_xor(m, o, 32));
    float s = 0.f;
    for (int k = lane; k < L; k += 32) s += __expf((float)row[k] - m);
#pragma unroll
    for (int o = 16; o; o >>= 1) s += __shfl_xor(s, o, 32);
    float inv = 1.f / s;
    for (int k = lane; k < L; k += 32) row[k] = (_Float16)(__expf((float)row[k] - m) * inv);
    for (int k = L + lane; k < L_pad; k += 32) row[k] = (_Float16)0.f;
}

// O = qh + A @ Vh (accumulator initialized with qh); grid (q_tiles*2, 4, 128), 32 thr
__global__ __launch_bounds__(32) void attn_av(const _Float16* __restrict__ S,
                                              const float* __restrict__ V,
                                              const float* __restrict__ Q,
                                              float* __restrict__ O,
                                              int nq, int nq_pad, int L_pad,
                                              int q_rpb, int v_rpb, int o_rpb) {
    int b = blockIdx.z, h = blockIdx.y;
    int qt = blockIdx.x >> 1, nt = blockIdx.x & 1;
    int lane = threadIdx.x, half = lane >> 4, l15 = lane & 15;
    int col = h * 32 + nt * 16 + l15;

    v8f acc;
#pragma unroll
    for (int v = 0; v < 8; ++v) {
        int m = qt * 16 + half * 8 + v;
        long qrow = (long)(q_rpb ? b * q_rpb : 0) + m;
        acc[v] = Q[qrow * 128 + col];
    }
    long sbase = ((long)(b * 4 + h)) * nq_pad * L_pad;
    const unsigned* srowu = (const unsigned*)(S + sbase + (long)(qt * 16 + l15) * L_pad);

    for (int kc = 0; kc < L_pad; kc += 32) {
        FragCast fa;
#pragma unroll
        for (int p = 0; p < 8; ++p) {
            int kb = half * 16 + 2 * p;
            fa.u[p] = srowu[(kc + kb) >> 1];
        }
        FragCast fb;
#pragma unroll
        for (int p = 0; p < 8; ++p) {
            int k0 = kc + half * 16 + 2 * p;
            long r0 = (long)b * v_rpb + k0;
            fb.u[p] = pack2h(V[r0 * 128 + col], V[(r0 + 1) * 128 + col]);
        }
        acc = __builtin_amdgcn_wmma_f32_16x16x32_f16(false, fa.h, false, fb.h,
                                                     (short)0, acc, false, false);
    }
#pragma unroll
    for (int v = 0; v < 8; ++v) {
        int m = qt * 16 + half * 8 + v;
        if (m < nq) O[((long)b * o_rpb + m) * 128 + col] = acc[v];
    }
}

// ---------------- host orchestration ----------------

extern "C" void kernel_launch(void* const* d_in, const int* in_sizes, int n_in,
                              void* d_out, int out_size, void* d_ws, size_t ws_size,
                              hipStream_t stream) {
    (void)in_sizes; (void)n_in; (void)out_size; (void)ws_size;
    const int N = 65536, E = 1048576, NG = 128;

    const float* x   = (const float*)d_in[0];
    const int* eidx  = (const int*)d_in[1];
    const int* src   = eidx;
    const int* dst   = eidx + E;
    const float *c1W = (const float*)d_in[3],  *c1b = (const float*)d_in[4];
    const float *c2W = (const float*)d_in[5],  *c2b = (const float*)d_in[6];
    const float *c3W = (const float*)d_in[7],  *c3b = (const float*)d_in[8];
    const float *lin1W = (const float*)d_in[9],  *lin1b = (const float*)d_in[10];
    const float *lin2W = (const float*)d_in[11], *lin2b = (const float*)d_in[12];
    const float *S1 = (const float*)d_in[13], *S2 = (const float*)d_in[14];
    const float *p1Wq = (const float*)d_in[15], *p1bq = (const float*)d_in[16];
    const float *p1Wk = (const float*)d_in[17], *p1bk = (const float*)d_in[18];
    const float *p1Wv = (const float*)d_in[19], *p1bv = (const float*)d_in[20];
    const float *p1Wo = (const float*)d_in[21], *p1bo = (const float*)d_in[22];
    const float *saWq = (const float*)d_in[23], *sabq = (const float*)d_in[24];
    const float *saWk = (const float*)d_in[25], *sabk = (const float*)d_in[26];
    const float *saWv = (const float*)d_in[27], *sabv = (const float*)d_in[28];
    const float *saWo = (const float*)d_in[29], *sabo = (const float*)d_in[30];
    const float *p2Wq = (const float*)d_in[31], *p2bq = (const float*)d_in[32];
    const float *p2Wk = (const float*)d_in[33], *p2bk = (const float*)d_in[34];
    const float *p2Wv = (const float*)d_in[35], *p2bv = (const float*)d_in[36];
    const float *p2Wo = (const float*)d_in[37], *p2bo = (const float*)d_in[38];
    const float *m1W = (const float*)d_in[39], *m1b = (const float*)d_in[40];
    const float *m2W = (const float*)d_in[41], *m2b = (const float*)d_in[42];

    // workspace carve-out
    size_t off = 0;
    auto carve = [&](size_t bytes) -> void* {
        void* p = (char*)d_ws + off;
        off += (bytes + 255) & ~(size_t)255;
        return p;
    };
    float* deg    = (float*)carve((size_t)N * 4);
    float* dinv   = (float*)carve((size_t)N * 4);
    float* invdeg = (float*)carve((size_t)N * 4);
    float* bufA   = (float*)carve((size_t)N * 128 * 4);
    float* bufB   = (float*)carve((size_t)N * 128 * 4);
    float* bufC   = (float*)carve((size_t)N * 128 * 4);
    float* bufD   = (float*)carve((size_t)N * 128 * 4);
    float* Qp     = (float*)carve((size_t)80 * 128 * 4);
    float* Qp2    = (float*)carve((size_t)16 * 128 * 4);
    _Float16* Sb  = (_Float16*)carve((size_t)NG * 4 * 80 * 512 * 2);
    const int RP = 9728;   // padded rows for the [128*75,128] class buffers
    float* O1 = (float*)carve((size_t)RP * 128 * 4);
    float* h1 = (float*)carve((size_t)RP * 128 * 4);
    float* t0 = (float*)carve((size_t)RP * 128 * 4);
    float* t1 = (float*)carve((size_t)RP * 128 * 4);
    float* t2 = (float*)carve((size_t)RP * 128 * 4);
    float* O2 = (float*)carve((size_t)RP * 128 * 4);
    float* h2 = (float*)carve((size_t)RP * 128 * 4);
    float* O3 = (float*)carve((size_t)128 * 128 * 4);
    float* g1 = (float*)carve((size_t)128 * 128 * 4);
    float* g2 = (float*)carve((size_t)128 * 128 * 4);
    float* g3 = (float*)carve((size_t)128 * 128 * 4);

    auto gemm = [&](const float* A, const float* W, const float* bias,
                    const float* skip, float* C, int M, int K, int mode) {
        gemm128<<<dim3((M + 63) / 64), 256, 0, stream>>>(A, W, bias, skip, C, M, K, mode);
    };
    auto zero = [&](float* p, long n) {
        zero_f32<<<dim3((unsigned)((n + 255) / 256)), 256, 0, stream>>>(p, n);
    };

    // degrees (shared by all 5 GCN convs)
    init_deg<<<N / 256, 256, 0, stream>>>(deg, N);
    count_deg<<<E / 256, 256, 0, stream>>>(dst, deg, E);
    finalize_deg<<<N / 256, 256, 0, stream>>>(deg, dinv, invdeg, N);

    auto gcn = [&](const float* in, int K, const float* W, const float* b,
                   float* tmp, float* out, int relu) {
        gemm(in, W, nullptr, nullptr, tmp, N, K, 0);
        zero(out, (long)N * 128);
        gcn_scatter<<<dim3((unsigned)(((long)E * 32) / 256)), 256, 0, stream>>>(
            tmp, src, dst, dinv, out, E);
        gcn_finalize<<<dim3((unsigned)(((long)N * 128) / 256)), 256, 0, stream>>>(
            out, tmp, invdeg, b, relu, (long)N * 128);
    };

    // three GCN layers with relu
    gcn(x,    256, c1W, c1b, bufA, bufB, 1);
    gcn(bufB, 128, c2W, c2b, bufA, bufC, 1);
    gcn(bufC, 128, c3W, c3b, bufA, bufB, 1);

    // lin1
    gemm(bufB, lin1W, lin1b, nullptr, bufC, N, 128, 0);      // x' = bufC
    // GMPool_G keys/values via GCNConv
    gcn(bufC, 128, p1Wk, p1bk, bufD, bufA, 0);               // K = bufA
    gcn(bufC, 128, p1Wv, p1bv, bufD, bufB, 0);               // V = bufB
    // shared query projection (S1 identical per graph)
    zero(Qp, 80 * 128);
    gemm(S1, p1Wq, p1bq, nullptr, Qp, 75, 128, 0);

    // MAB 1: nq=75(pad80), L=512
    attn_scores<<<dim3(5 * 32, 4, NG), 32, 0, stream>>>(Qp, bufA, Sb, 80, 512, 0, 512, 32);
    attn_softmax<<<dim3(75, 4, NG), 32, 0, stream>>>(Sb, 80, 512, 512);
    attn_av<<<dim3(10, 4, NG), 32, 0, stream>>>(Sb, bufB, Qp, O1, 75, 80, 512, 0, 512, 75);
    gemm(O1, p1Wo, p1bo, O1, h1, 9600, 128, 2);

    // MAB 2 (SAB): nq=75, L=75(pad96)
    gemm(h1, saWq, sabq, nullptr, t0, 9600, 128, 0);
    gemm(h1, saWk, sabk, nullptr, t1, 9600, 128, 0);
    gemm(h1, saWv, sabv, nullptr, t2, 9600, 128, 0);
    attn_scores<<<dim3(5 * 5, 4, NG), 32, 0, stream>>>(t0, t1, Sb, 80, 96, 75, 75, 5);
    attn_softmax<<<dim3(75, 4, NG), 32, 0, stream>>>(Sb, 80, 75, 96);
    attn_av<<<dim3(10, 4, NG), 32, 0, stream>>>(Sb, t2, t0, O2, 75, 80, 96, 75, 75, 75);
    gemm(O2, saWo, sabo, O2, h2, 9600, 128, 2);

    // MAB 3 (GMPool_I): nq=1(pad16), L=75(pad96)
    zero(Qp2, 16 * 128);
    gemm(S2, p2Wq, p2bq, nullptr, Qp2, 1, 128, 0);
    gemm(h2, p2Wk, p2bk, nullptr, t0, 9600, 128, 0);
    gemm(h2, p2Wv, p2bv, nullptr, t1, 9600, 128, 0);
    attn_scores<<<dim3(1 * 5, 4, NG), 32, 0, stream>>>(Qp2, t0, Sb, 16, 96, 0, 75, 5);
    attn_softmax<<<dim3(1, 4, NG), 32, 0, stream>>>(Sb, 16, 75, 96);
    attn_av<<<dim3(2, 4, NG), 32, 0, stream>>>(Sb, t1, Qp2, O3, 1, 16, 96, 0, 75, 1);
    gemm(O3, p2Wo, p2bo, O3, g1, 128, 128, 2);

    // head MLP
    gemm(g1, lin2W, lin2b, nullptr, g2, 128, 128, 0);
    gemm(g2, m1W, m1b, nullptr, g3, 128, 128, 1);
    gemm(g3, m2W, m2b, nullptr, (float*)d_out, 128, 128, 0);
}
